// DRCB_12017318494352
// MI455X (gfx1250) — matrix operand order repaired and checked by reading, not compiled, not run on hardware
//
#include <hip/hip_runtime.h>

typedef __attribute__((ext_vector_type(16))) _Float16 v16h;
typedef __attribute__((ext_vector_type(8)))  _Float16 v8h;
typedef __attribute__((ext_vector_type(8)))  float    v8f;

#define B_ 16
#define C_ 64
#define H_ 192
#define W_ 192
#define E_ 3
#define HW_ (H_*W_)

#define COLS 66   // 64 px + 2 halo
#define ROWS 4    // 2 rows + 2 halo
#define CSTR 72   // padded cin stride in halves (multiple of 8 -> 16B-aligned b128)

// ---------------- Phase 1: per-(b,c) plane mean ----------------
__global__ __launch_bounds__(256) void pool_kernel(const float* __restrict__ x,
                                                   float* __restrict__ pooled) {
    int bc = blockIdx.x;                       // 0..B*C-1
    const float* p = x + (size_t)bc * HW_;
    float s = 0.f;
    for (int i = threadIdx.x; i < HW_; i += 256) s += p[i];
    __shared__ float red[256];
    red[threadIdx.x] = s;
    __syncthreads();
    for (int off = 128; off > 0; off >>= 1) {
        if (threadIdx.x < off) red[threadIdx.x] += red[threadIdx.x + off];
        __syncthreads();
    }
    if (threadIdx.x == 0) pooled[bc] = red[0] * (1.0f / (float)HW_);
}

// ---------------- Phase 2: routing softmax(logits/30) ----------------
__global__ void route_kernel(const float* __restrict__ pooled,
                             const float* __restrict__ fc_w,
                             const float* __restrict__ fc_b,
                             float* __restrict__ r) {
    int b = threadIdx.x;
    if (b >= B_) return;
    float lg[E_];
    float mx = -1e30f;
    for (int e = 0; e < E_; ++e) {
        float s = fc_b[e];
        for (int c = 0; c < C_; ++c) s += pooled[b*C_ + c] * fc_w[e*C_ + c];
        lg[e] = s * (1.0f / 30.0f);
        mx = fmaxf(mx, lg[e]);
    }
    float den = 0.f;
    for (int e = 0; e < E_; ++e) { lg[e] = __expf(lg[e] - mx); den += lg[e]; }
    for (int e = 0; e < E_; ++e) r[b*E_ + e] = lg[e] / den;
}

// ---------------- Phase 3: mix experts -> combined f16 3x3 kernel ----------------
// Wc layout: [b][tap][cout][cin] f16.  tap4 (center) absorbs the 1x1 branch + identity.
__global__ __launch_bounds__(256) void mix_kernel(const float* __restrict__ w3,
                                                  const float* __restrict__ w1,
                                                  const float* __restrict__ r,
                                                  _Float16* __restrict__ wc) {
    int b   = blockIdx.x / 9;
    int tap = blockIdx.x % 9;
    float r0 = r[b*E_ + 0], r1 = r[b*E_ + 1], r2 = r[b*E_ + 2];
    for (int idx = threadIdx.x; idx < C_*C_; idx += 256) {
        int m   = idx >> 6;
        int cin = idx & 63;
        int base = (m*C_ + cin)*9 + tap;               // w3[e][o][i][kh][kw]
        float v = r0*w3[base] + r1*w3[C_*C_*9 + base] + r2*w3[2*C_*C_*9 + base];
        if (tap == 4) {                                // center tap: + 1x1 + identity
            int b1 = m*C_ + cin;
            v += r0*w1[b1] + r1*w1[C_*C_ + b1] + r2*w1[2*C_*C_ + b1];
            if (m == cin) v += 1.0f;
        }
        wc[(((size_t)b*9 + tap)*C_ + m)*C_ + cin] = (_Float16)v;
    }
}

// ---------------- Phase 4: implicit-GEMM conv3x3 + ReLU via WMMA ----------------
// Block: 64 cout x (2 rows x 64 px). 8 waves, each owns one M-tile (16 cout) and
// 4 N-tiles (16 px each). K = tap*64 + cin, stepped in groups of 32 (fixed tap).
__global__ __launch_bounds__(256) void conv_kernel(const float* __restrict__ x,
                                                   const _Float16* __restrict__ wc,
                                                   float* __restrict__ out) {
    __shared__ _Float16 smem[ROWS * COLS * CSTR];     // [row][col][cin], padded
    int bx  = blockIdx.x;            // 0..2   : 64-px x-tile
    int by  = blockIdx.y;            // 0..95  : row pair
    int b   = blockIdx.z;            // batch
    int px0 = bx * 64;
    int y0  = by * 2;
    int t   = threadIdx.x;

    // ---- stage x halo tile (4 rows x 66 cols x 64 cin) as f16, zero-padded ----
    const float* xb = x + (size_t)b * C_ * HW_;
    for (int idx = t; idx < C_ * ROWS * COLS; idx += 256) {
        int col = idx % COLS;        // fastest -> coalesced global reads along W
        int tmp = idx / COLS;
        int rr  = tmp & 3;
        int cin = tmp >> 2;
        int y   = y0 - 1 + rr;
        int xc  = px0 - 1 + col;
        float v = 0.f;
        if ((unsigned)y < (unsigned)H_ && (unsigned)xc < (unsigned)W_)
            v = xb[(size_t)cin * HW_ + (size_t)y * W_ + xc];
        smem[(rr * COLS + col) * CSTR + cin] = (_Float16)v;
    }
    __syncthreads();

    int w    = t >> 5;
    int lane = t & 31;
    int mt   = w & 3;                // M-tile (16 cout)
    int ntb  = (w >> 2) * 4;         // first of 4 N-tiles
    int lo16 = lane & 15;
    int hi   = lane >> 4;            // lane half-select (contiguous K split: k = 16*hi + 0..15)

    v8f acc0 = {}, acc1 = {}, acc2 = {}, acc3 = {};

    // per-lane A base: Wc[b][tap][m][cin], m fixed per lane
    const _Float16* wcb = wc + (size_t)b * 9 * C_ * C_ + (size_t)(mt*16 + lo16) * C_;

    for (int g = 0; g < 18; ++g) {   // 9 taps x 2 halves of cin
        int tap = g >> 1;
        int c0  = (g & 1) * 32;
        int ky  = tap / 3;
        int kx  = tap % 3;

        // A fragment: 16x32 f16. K = 16*hi + 0..15 = contiguous cin run (matches B).
        const _Float16* ap = wcb + (size_t)tap * (C_*C_) + c0 + 16*hi;
        v8h a_lo = *(const v8h*)ap;          // k_local = 16*hi + 0..7
        v8h a_hi = *(const v8h*)(ap + 8);    // k_local = 16*hi + 8..15
        v16h a;
        #pragma unroll
        for (int i = 0; i < 8; ++i) { a[i] = a_lo[i]; a[i+8] = a_hi[i]; }

        #pragma unroll
        for (int j = 0; j < 4; ++j) {
            int nt = ntb + j;
            int rr = (nt >> 2) + ky;                 // LDS row (halo offset folded)
            int cc = (nt & 3) * 16 + lo16 + kx;      // LDS col
            // B fragment: 32x16 f16, lane = N, K = 16*hi + 0..15 = contiguous cin run
            const _Float16* bp = &smem[(rr * COLS + cc) * CSTR + c0 + 16*hi];
            v8h b_lo = *(const v8h*)bp;
            v8h b_hi = *(const v8h*)(bp + 8);
            v16h bf;
            #pragma unroll
            for (int i = 0; i < 8; ++i) { bf[i] = b_lo[i]; bf[i+8] = b_hi[i]; }

            v8f* accp = (j == 0) ? &acc0 : (j == 1) ? &acc1 : (j == 2) ? &acc2 : &acc3;
            *accp = __builtin_amdgcn_wmma_f32_16x16x32_f16(
                false, a, false, bf, (short)0, *accp, false, false);
        }
    }

    // ---- epilogue: ReLU + store (identity already folded into weights) ----
    float* ob = out + (size_t)b * C_ * HW_;
    #pragma unroll
    for (int j = 0; j < 4; ++j) {
        int nt = ntb + j;
        int y  = y0 + (nt >> 2);
        int xg = px0 + (nt & 3) * 16 + lo16;
        v8f acc = (j == 0) ? acc0 : (j == 1) ? acc1 : (j == 2) ? acc2 : acc3;
        #pragma unroll
        for (int v = 0; v < 8; ++v) {
            int m = mt*16 + v + 8*hi;               // C/D layout: VGPR v -> M row
            ob[(size_t)m * HW_ + (size_t)y * W_ + xg] = fmaxf(acc[v], 0.f);
        }
    }
}

extern "C" void kernel_launch(void* const* d_in, const int* in_sizes, int n_in,
                              void* d_out, int out_size, void* d_ws, size_t ws_size,
                              hipStream_t stream) {
    const float* x    = (const float*)d_in[0];   // [16,64,192,192]
    const float* w3   = (const float*)d_in[1];   // [3,64,64,3,3]
    const float* w1   = (const float*)d_in[2];   // [3,64,64,1,1]
    const float* fc_w = (const float*)d_in[3];   // [3,64]
    const float* fc_b = (const float*)d_in[4];   // [3]
    float* out = (float*)d_out;

    char* ws = (char*)d_ws;
    float*    pooled = (float*)ws;               // 1024 floats
    float*    r      = (float*)(ws + 4096);      // 48 floats
    _Float16* wcmb   = (_Float16*)(ws + 8192);   // 16*9*64*64 halves (~1.13 MB)

    pool_kernel <<<B_*C_, 256, 0, stream>>>(x, pooled);
    route_kernel<<<1, 32, 0, stream>>>(pooled, fc_w, fc_b, r);
    mix_kernel  <<<B_*9, 256, 0, stream>>>(w3, w1, r, wcmb);

    dim3 grid(W_/64, H_/2, B_);                  // (3, 96, 16)
    conv_kernel <<<grid, 256, 0, stream>>>(x, wcmb, out);
}
